// KernelDensityEstimation_14645838479442
// MI455X (gfx1250) — compile-verified, matrix operand order checked by reading, not run on hardware
//
#include <hip/hip_runtime.h>
#include <hip/hip_bf16.h>
#include <math.h>

typedef __attribute__((ext_vector_type(16))) _Float16 v16h;
typedef __attribute__((ext_vector_type(8)))  float    v8f;

#define NROWS 512
#define NX    256
#define NQ    2048
#define RG    1024
#define INV_C 0.015583684f   /* 1/(256*0.1*sqrt(2*pi)) */

// ---------------------------------------------------------------------------
// Kernel A: per-row min/max of samples s and transformed y.
// One block per row, 256 threads, x-column staged in LDS.
// ---------------------------------------------------------------------------
__global__ __launch_bounds__(256) void row_stats_kernel(
    const float* __restrict__ inputs, const float* __restrict__ noise,
    const int* __restrict__ comp_idx, const int* __restrict__ sel,
    float* __restrict__ stats)
{
    const int r = blockIdx.x;
    const int tid = threadIdx.x;
    __shared__ float xs[NX];
    __shared__ float red0[256], red1[256], red2[256], red3[256];

    xs[tid] = inputs[(size_t)tid * NROWS + r];   // x[r, tid] = inputs[tid, r]
    __syncthreads();

    const int sr = sel[r];
    float smin = 3.4e38f, smax = -3.4e38f, ymin = 3.4e38f, ymax = -3.4e38f;

    for (int q = tid; q < NQ; q += 256) {
        const int ci = comp_idx[(size_t)r * NQ + q];
        const float s = xs[ci] + 0.1f * noise[(size_t)r * NQ + q];
        smin = fminf(smin, s); smax = fmaxf(smax, s);
        float y;
        switch (sr) {
            case 0: y = 2.0f * s; break;
            case 1: y = s - 3.0f; break;
            case 2: { float ss = (fabsf(s) < 1e-6f) ? 1e-6f : s; y = 1.0f / ss; } break;
            case 3: y = 0.8f * s + 5.0f; break;
            case 4: y = 1.0f / (1.0f + __expf(-s)); break;
            default: y = fmaxf(s, 0.0f) + log1pf(__expf(-fabsf(s))); break; // softplus
        }
        ymin = fminf(ymin, y); ymax = fmaxf(ymax, y);
    }

    red0[tid] = smin; red1[tid] = smax; red2[tid] = ymin; red3[tid] = ymax;
    __syncthreads();
    for (int ofs = 128; ofs > 0; ofs >>= 1) {
        if (tid < ofs) {
            red0[tid] = fminf(red0[tid], red0[tid + ofs]);
            red1[tid] = fmaxf(red1[tid], red1[tid + ofs]);
            red2[tid] = fminf(red2[tid], red2[tid + ofs]);
            red3[tid] = fmaxf(red3[tid], red3[tid + ofs]);
        }
        __syncthreads();
    }
    if (tid == 0) {
        stats[r * 4 + 0] = red0[0]; stats[r * 4 + 1] = red1[0];
        stats[r * 4 + 2] = red2[0]; stats[r * 4 + 3] = red3[0];
    }
}

// ---------------------------------------------------------------------------
// Kernel B: KDE evaluation. One block per row (8 waves). Each wave processes
// groups of 16 grid points; the 256-term j-reduction per point is done as
// 8x v_wmma_f32_16x16x32_f16 with B == all-ones (D[M,*] = row sums, f32 acc).
// Log-det ld is folded into each term's exponent: term = exp(ld - 0.5*z^2).
// Writes slots 0 (current) and 1 (transformed) of the output bank directly.
// ---------------------------------------------------------------------------
__global__ __launch_bounds__(256) void kde_wmma_kernel(
    const float* __restrict__ inputs, const int* __restrict__ sel,
    const float* __restrict__ stats, float* __restrict__ out)
{
    const int r = blockIdx.x;
    const int tid = threadIdx.x;
    __shared__ float xs[NX];
    xs[tid] = inputs[(size_t)tid * NROWS + r];
    __syncthreads();

    const float smin = stats[r * 4 + 0], smax = stats[r * 4 + 1];
    const float ymin = stats[r * 4 + 2], ymax = stats[r * 4 + 3];
    const int sr = sel[r];

    const int wave = tid >> 5;
    const int lane = tid & 31;
    const int M = lane & 15;     // A row (C/D column) within 16-group (ISA layout)
    const int hi = lane >> 4;

    v16h bones;
#pragma unroll
    for (int h = 0; h < 16; ++h) bones[h] = (_Float16)1.0f;

    // 128 groups of 16 points: g<64 -> "current" (t0 grid), g>=64 -> "transformed".
    for (int g = wave; g < 128; g += 8) {
        const int target = g >> 6;
        const int base16 = (g & 63) * 16;
        const int i = base16 + M;
        const float u = (float)i / 1023.0f;

        float loc, ld;
        if (target == 0) {
            float t = smin + (smax - smin) * u;
            if (t == 0.0f) t = 1e-7f;
            loc = t; ld = 0.0f;
        } else {
            float t = ymin + (ymax - ymin) * u;
            if (t == 0.0f) t = 1e-7f;
            switch (sr) {
                case 0: loc = t * 0.5f;           ld = -0.6931471805599453f; break;
                case 1: loc = t + 3.0f;           ld = 0.0f;                 break;
                case 2: loc = 1.0f / t;           ld = -2.0f * logf(fabsf(t)); break;
                case 3: loc = (t - 5.0f) * 1.25f; ld = 0.22314355131420976f; break;
                case 4: {
                    float yc = fminf(fmaxf(t, 1e-6f), 1.0f - 1e-6f);
                    float la = logf(yc), lb = log1pf(-yc);
                    loc = la - lb; ld = -la - lb;
                } break;
                default: {
                    float yc = fmaxf(t, 1e-6f);
                    float lem = logf(expm1f(yc));
                    loc = lem; ld = yc - lem;
                } break;
            }
        }

        v8f acc = {0.f, 0.f, 0.f, 0.f, 0.f, 0.f, 0.f, 0.f};
        for (int kk = 0; kk < 8; ++kk) {
            v16h a;
#pragma unroll
            for (int h = 0; h < 16; ++h) {
                // ISA 16-bit A layout: lanes 0-15 hold K={0..7,16..23},
                // lanes 16-31 hold K={8..15,24..31}. With B==1 any in-lane
                // K permutation yields the same row sum.
                const int K = ((h < 8) ? h : h + 8) + hi * 8;
                const float xv = xs[kk * 32 + K];
                const float z = (loc - xv) * 10.0f;               // /H, H=0.1
                const float e = __expf(fmaf(-0.5f * z, z, ld));
                a[h] = (_Float16)e;
            }
            acc = __builtin_amdgcn_wmma_f32_16x16x32_f16(
                false, a, false, bones, (short)0, acc, false, false);
        }

        // D layout: lanes 0-15 hold rows M=v, lanes 16-31 rows M=v+8; all
        // columns are equal (B==1), so lanes 0 and 16 carry all 16 row sums.
        if (M == 0) {
            float* dst = out + (size_t)target * (NROWS * RG)
                             + (size_t)r * RG + base16 + hi * 8;
            float4 lo = make_float4(acc[0] * INV_C, acc[1] * INV_C,
                                    acc[2] * INV_C, acc[3] * INV_C);
            float4 hl = make_float4(acc[4] * INV_C, acc[5] * INV_C,
                                    acc[6] * INV_C, acc[7] * INV_C);
            *(float4*)(dst + 0) = lo;
            *(float4*)(dst + 4) = hl;
        }
    }
}

// ---------------------------------------------------------------------------
// Kernel C: bandwidth-bound slot copy (slots 2..127), with out[slot_idx] =
// bank[0]. float4 (global_load_b128/global_store_b128) grid-stride.
// ---------------------------------------------------------------------------
__global__ __launch_bounds__(256) void copy_slots_kernel(
    const float* __restrict__ bank, const int* __restrict__ slot_idx,
    float* __restrict__ out)
{
    const int F4_PER_SLOT = NROWS * RG / 4;           // 131072 = 1<<17
    const long long n = 126LL * F4_PER_SLOT;
    const int sl = slot_idx[0];
    const float4* src4 = (const float4*)bank;
    float4* dst4 = (float4*)out;
    for (long long idx = (long long)blockIdx.x * blockDim.x + threadIdx.x;
         idx < n; idx += (long long)gridDim.x * blockDim.x) {
        const int slot = 2 + (int)(idx >> 17);
        const int within = (int)(idx & (F4_PER_SLOT - 1));
        const int src = (slot == sl) ? 0 : slot;
        dst4[(size_t)slot * F4_PER_SLOT + within] =
            src4[(size_t)src * F4_PER_SLOT + within];
    }
}

extern "C" void kernel_launch(void* const* d_in, const int* in_sizes, int n_in,
                              void* d_out, int out_size, void* d_ws, size_t ws_size,
                              hipStream_t stream)
{
    const float* inputs   = (const float*)d_in[0];   // (256, 512) f32
    const float* bank     = (const float*)d_in[1];   // (128, 512, 1024) f32
    const float* noise    = (const float*)d_in[2];   // (512, 2048) f32
    const int*   comp_idx = (const int*)d_in[3];     // (512, 2048) i32
    const int*   sel      = (const int*)d_in[4];     // (512,) i32
    const int*   slot_idx = (const int*)d_in[5];     // scalar i32 (37)
    float* out   = (float*)d_out;
    float* stats = (float*)d_ws;                     // 512*4 floats

    row_stats_kernel<<<dim3(NROWS), dim3(256), 0, stream>>>(
        inputs, noise, comp_idx, sel, stats);
    kde_wmma_kernel<<<dim3(NROWS), dim3(256), 0, stream>>>(
        inputs, sel, stats, out);
    copy_slots_kernel<<<dim3(8192), dim3(256), 0, stream>>>(
        bank, slot_idx, out);
}
